// PeakBinner_68968584839264
// MI455X (gfx1250) — matrix operand order repaired and checked by reading, not compiled, not run on hardware
//
#include <hip/hip_runtime.h>

// ---- problem constants ----
#define B_ROWS     4096
#define INPUT_SIZE 10000
#define GROUP_SIZE 500
#define STEP       250
#define OUT_PER_G  50
#define GROUPS     39
#define OUT_W      (GROUPS * OUT_PER_G)   // 1950

// ---- tiling ----
#define MBLK       64                      // rows per block (4 waves x 16)
#define KCHUNK     256                     // halves of K staged per LDS chunk
#define LDSS       264                     // LDS row stride in halves (256 + 8 pad)

typedef _Float16 v16h __attribute__((ext_vector_type(16)));
typedef _Float16 h8   __attribute__((ext_vector_type(8)));
typedef _Float16 h2   __attribute__((ext_vector_type(2)));
typedef float    v8f  __attribute__((ext_vector_type(8)));

__device__ __forceinline__ v16h frag16(const _Float16* p0, const _Float16* p1) {
    h8 lo = *(const h8*)p0;
    h8 hi = *(const h8*)p1;
    return __builtin_shufflevector(lo, hi, 0,1,2,3,4,5,6,7,8,9,10,11,12,13,14,15);
}

__global__ __launch_bounds__(128, 4)
void peakbinner_wmma_kernel(const float* __restrict__ x,
                            const float* __restrict__ W,
                            float* __restrict__ out) {
    __shared__ _Float16 xs[MBLK * LDSS];       // 64 x 264 halves
    __shared__ _Float16 ws[OUT_PER_G * LDSS];  // 50 x 264 halves

    const int tid  = threadIdx.x;
    const int g    = blockIdx.x;          // group 0..38
    const int m0   = blockIdx.y * MBLK;   // row block

    const float* __restrict__ xg = x + (size_t)m0 * INPUT_SIZE + g * STEP;
    const float* __restrict__ wg = W + (size_t)g * (OUT_PER_G * GROUP_SIZE);

    const int lane = tid & 31;
    const int wv   = tid >> 5;            // wave 0..3 -> M-tile
    const int hsel = lane >> 4;           // half-wave select
    const int lr   = lane & 15;

    // per-wave LDS row bases (chunk-invariant)
    const _Float16* arow  = xs + (wv * 16 + lr) * LDSS;        // A: row = M
    const _Float16* brow0 = ws + (lr)       * LDSS;            // B tiles: row = N
    const _Float16* brow1 = ws + (16 + lr)  * LDSS;
    const _Float16* brow2 = ws + (32 + lr)  * LDSS;
    const int r3 = (48 + lr) < OUT_PER_G ? (48 + lr) : (OUT_PER_G - 1);
    const _Float16* brow3 = ws + r3 * LDSS;                    // clamped (cols >=50 discarded)

    v8f acc[4] = {v8f{}, v8f{}, v8f{}, v8f{}};

    // Each thread owns one fixed column pair: k = 2*tid covers all 256 halves
    // of a chunk with 128 threads. Row loops below are uniform, constant-trip,
    // fully independent -> batched global_load_b64 with immediate offsets.
    const int kth = tid << 1;             // 0..254 (this thread's k within chunk)

    #pragma unroll
    for (int kc = 0; kc < 2; ++kc) {
        const int koff   = kc * KCHUNK;
        const int kvalid = GROUP_SIZE - koff;   // 500 (chunk0) or 244 (chunk1)
        const bool ok    = (kth < kvalid);      // chunk0: always true (folds away)
        const int  kcl   = ok ? kth : 0;        // clamped column, always in-bounds

        // ---- stage x chunk: 64 rows, this thread's column pair ----
        {
            const float2*  src = (const float2*)(xg + koff + kcl);
            _Float16*      dst = xs + kth;
            #pragma unroll 8
            for (int row = 0; row < MBLK; ++row) {
                float2 v = src[row * (INPUT_SIZE / 2)];   // const offset: row*40000 B
                h2 h;
                h.x = (_Float16)(ok ? v.x : 0.f);
                h.y = (_Float16)(ok ? v.y : 0.f);
                *(h2*)(dst + row * LDSS) = h;             // const offset: row*528 B
            }
        }
        // ---- stage W chunk: 50 rows, this thread's column pair ----
        {
            const float2*  src = (const float2*)(wg + koff + kcl);
            _Float16*      dst = ws + kth;
            #pragma unroll 8
            for (int row = 0; row < OUT_PER_G; ++row) {
                float2 v = src[row * (GROUP_SIZE / 2)];   // const offset: row*2000 B
                h2 h;
                h.x = (_Float16)(ok ? v.x : 0.f);
                h.y = (_Float16)(ok ? v.y : 0.f);
                *(h2*)(dst + row * LDSS) = h;
            }
        }
        __syncthreads();

        // ---- 8 K-steps of 32 per chunk; unroll 2 keeps VGPRs < 256 while
        // still overlapping next step's ds_load_b128 with current WMMAs ----
        #pragma unroll 2
        for (int kt = 0; kt < KCHUNK / 32; ++kt) {
            const int kb = kt * 32;
            // A fragment: ISA A-layout (lanes<16: K 0..7 & 16..23; lanes>=16: +8)
            const int ao = kb + hsel * 8;
            v16h A = frag16(arow + ao, arow + ao + 16);
            // B fragments: same K layout as A (lanes<16: K 0..7 & 16..23; lanes>=16: +8)
            const int bo = kb + hsel * 8;
            v16h B0 = frag16(brow0 + bo, brow0 + bo + 16);
            v16h B1 = frag16(brow1 + bo, brow1 + bo + 16);
            v16h B2 = frag16(brow2 + bo, brow2 + bo + 16);
            v16h B3 = frag16(brow3 + bo, brow3 + bo + 16);

            acc[0] = __builtin_amdgcn_wmma_f32_16x16x32_f16(false, A, false, B0, (short)0, acc[0], false, false);
            acc[1] = __builtin_amdgcn_wmma_f32_16x16x32_f16(false, A, false, B1, (short)0, acc[1], false, false);
            acc[2] = __builtin_amdgcn_wmma_f32_16x16x32_f16(false, A, false, B2, (short)0, acc[2], false, false);
            acc[3] = __builtin_amdgcn_wmma_f32_16x16x32_f16(false, A, false, B3, (short)0, acc[3], false, false);
        }
        __syncthreads();
    }

    // ---- ReLU + store. D layout: VGPR r -> M = r + 8*hsel; lane%16 -> N ----
    const size_t mbase = (size_t)(m0 + wv * 16 + hsel * 8);
    #pragma unroll
    for (int nt = 0; nt < 4; ++nt) {
        const int n = nt * 16 + lr;
        if (n < OUT_PER_G) {
            float* op = out + mbase * OUT_W + (size_t)g * OUT_PER_G + n;
            #pragma unroll
            for (int r = 0; r < 8; ++r) {
                float v = acc[nt][r];
                op[(size_t)r * OUT_W] = v > 0.f ? v : 0.f;
            }
        }
    }
}

extern "C" void kernel_launch(void* const* d_in, const int* in_sizes, int n_in,
                              void* d_out, int out_size, void* d_ws, size_t ws_size,
                              hipStream_t stream) {
    const float* x = (const float*)d_in[0];   // (4096, 10000) f32
    const float* W = (const float*)d_in[1];   // (39, 50, 500) f32
    float* out = (float*)d_out;               // (4096, 1950) f32

    dim3 grid(GROUPS, B_ROWS / MBLK, 1);      // 39 x 64 blocks
    dim3 block(128, 1, 1);                    // 4 waves (wave32)
    peakbinner_wmma_kernel<<<grid, block, 0, stream>>>(x, W, out);
}
